// GRU_32564442038643
// MI455X (gfx1250) — matrix operand order repaired and checked by reading, not compiled, not run on hardware
//
#include <hip/hip_runtime.h>

#define HID   256
#define INPD  256
#define DTOT  512   // INPD + HID
#define SEQL  512
#define BATCH 256
#define BT    16    // batch rows per workgroup

typedef __attribute__((ext_vector_type(16))) _Float16 v16h;
typedef __attribute__((ext_vector_type(8)))  float    v8f;

// ---------------------------------------------------------------------------
// WMMA fragment loaders (CDNA5 ISA 7.12.2 layouts, wave32)
// A 16x32 f16: lane L<16 -> M=L, K = {0..7} in u[0..3], {16..23} in u[4..7];
//              lane L>=16 -> M=L-16, K = {8..15} and {24..31}. Pairs per dword.
// rowPtr points at this lane's M row (already offset by k0).
__device__ __forceinline__ v16h fragA(const _Float16* rowPtr, int lane) {
  union { v16h v; unsigned u[8]; } f;
  const unsigned* p = (const unsigned*)rowPtr + ((lane & 16) ? 4 : 0);
#pragma unroll
  for (int j = 0; j < 4; ++j) { f.u[j] = p[j]; f.u[4 + j] = p[8 + j]; }
  return f.v;
}

// B 32x16 f16: lane = N column; lanes 0-15 hold K={0..7,16..23}, lanes 16-31
// hold K={8..15,24..31} (same order as A). rowPtr points at this lane's N row
// (weights stored transposed [N][K]), already offset by k0.
__device__ __forceinline__ v16h fragB(const _Float16* rowPtr, int lane) {
  union { v16h v; unsigned u[8]; } f;
  const unsigned* p = (const unsigned*)rowPtr + ((lane & 16) ? 4 : 0);
#pragma unroll
  for (int j = 0; j < 4; ++j) { f.u[j] = p[j]; f.u[4 + j] = p[8 + j]; }
  return f.v;
}

__device__ __forceinline__ v8f wmma_f16(v16h a, v16h b, v8f c) {
  return __builtin_amdgcn_wmma_f32_16x16x32_f16(false, a, false, b, (short)0, c,
                                                false, false);
}

__device__ __forceinline__ float fsig(float v) { return 1.f / (1.f + __expf(-v)); }
__device__ __forceinline__ float ftanh(float v) {
  float e = __expf(2.f * v);
  return (e - 1.f) / (e + 1.f);
}

// Opaque zero (new SSA value each call): adding it to a weight base defeats
// LICM (loads stay in the sequence loop, streamed from L2) while PRESERVING
// the base pointer's global address-space provenance, so the compiler emits
// global_load_b128 (LOADcnt only, SADDR form) instead of flat_load_b128
// (which would also consume DScnt and couple against the LDS A-stream).
__device__ __forceinline__ size_t opaque_zero() {
  unsigned long long v = 0;
  asm volatile("" : "+s"(v));
  return (size_t)v;
}

// ---------------------------------------------------------------------------
// Prep: convert fp32 weights to f16 and transpose to [N][K] for contiguous
// B-fragment loads.  Wt layout: Wr_t[256][512] | Wz_t[256][512] | Wh_t[256][512]
// | Wo_t[256][256]
__global__ void gru_pack_weights(const float* __restrict__ Wr,
                                 const float* __restrict__ Wz,
                                 const float* __restrict__ Wh,
                                 const float* __restrict__ Wo,
                                 _Float16* __restrict__ Wt) {
  const int GSZ = DTOT * HID;  // 131072
  int idx = blockIdx.x * blockDim.x + threadIdx.x;
  if (idx < 3 * GSZ) {
    int g = idx / GSZ, r = idx - g * GSZ;
    int k = r >> 8, n = r & 255;  // r = k*256 + n
    const float* W = (g == 0) ? Wr : (g == 1) ? Wz : Wh;
    Wt[(size_t)g * GSZ + (size_t)n * DTOT + k] = (_Float16)W[r];
  } else {
    int r = idx - 3 * GSZ;
    if (r < HID * HID) {
      int k = r >> 8, n = r & 255;
      Wt[(size_t)3 * GSZ + (size_t)n * HID + k] = (_Float16)Wo[r];
    }
  }
}

// ---------------------------------------------------------------------------
// Sequential GRU: one workgroup = 16 batch rows, 8 waves; each wave owns two
// 16-wide N tiles of the 256-wide hidden dimension. h lives in LDS (the upper
// half of `comb`) for the whole sequence; weights stream from L2 each step.
__global__ __launch_bounds__(256) void gru_seq_kernel(
    const float* __restrict__ x, const float* __restrict__ br,
    const float* __restrict__ bz, const float* __restrict__ bh,
    const float* __restrict__ bo, const _Float16* __restrict__ Wt,
    float* __restrict__ out) {
  __shared__ _Float16 comb[BT * DTOT];  // [16][512] = [x_t | h]   (16 KB)
  __shared__ _Float16 rh[BT * HID];     // [16][256] = r * h        (8 KB)

  const int tid = threadIdx.x;
  const int wave = tid >> 5;
  const int lane = tid & 31;
  const int bBase = blockIdx.x * BT;

  const _Float16* WrT = Wt;
  const _Float16* WzT = Wt + (size_t)DTOT * HID;
  const _Float16* WhT = Wt + (size_t)2 * DTOT * HID;
  const _Float16* WoT = Wt + (size_t)3 * DTOT * HID;

  const int nA = wave * 32 + (lane & 15);
  const int nB = nA + 16;
  const int mOff = (lane & 16) ? 8 : 0;

  // Loop-invariant per-lane offsets (elements) into the [N][K] weight arrays
  const size_t offA = (size_t)nA * DTOT;
  const size_t offB = (size_t)nB * DTOT;

  // A-row pointers (LDS)
  const _Float16* arow  = comb + (lane & 15) * DTOT;  // [x|h]
  const _Float16* arow2 = rh + (lane & 15) * HID;     // r*h

  const float brA = br[nA], brB = br[nB];
  const float bzA = bz[nA], bzB = bz[nB];
  const float bhA = bh[nA], bhB = bh[nB];

  // h0 = 0
  for (int i = tid; i < BT * HID; i += 256)
    comb[(i >> 8) * DTOT + INPD + (i & 255)] = (_Float16)0.f;
  __syncthreads();

#pragma unroll 1
  for (int t = 0; t < SEQL; ++t) {
    // ---- stage x_t tile into x-half of comb (fp32 -> f16)
    {
      int row = tid >> 4;
      int cb = (tid & 15) * 16;
      const float* xp = x + ((size_t)(bBase + row) * SEQL + t) * INPD + cb;
      _Float16* dst = comb + row * DTOT + cb;
#pragma unroll
      for (int j = 0; j < 16; ++j) dst[j] = (_Float16)xp[j];
      // warm L2/L0 for the next timestep's x tile (global_prefetch_b8)
      if (t + 1 < SEQL) __builtin_prefetch(xp + INPD, 0, 1);
    }
    __syncthreads();

    // Opaque per-iteration offset: weight loads below cannot be hoisted but
    // keep global address-space provenance.
    {
      const size_t oz = opaque_zero();
      const _Float16* WrB = WrT + oz;
      const _Float16* WzB = WzT + oz;

      // ---- r / z gates: [16x512] @ W{r,z}[512x256], two N tiles per wave
      v8f aR0 = {}, aR1 = {}, aZ0 = {}, aZ1 = {};
#pragma unroll
      for (int kk = 0; kk < DTOT; kk += 32) {
        v16h a = fragA(arow + kk, lane);
        aR0 = wmma_f16(a, fragB(WrB + offA + kk, lane), aR0);
        aR1 = wmma_f16(a, fragB(WrB + offB + kk, lane), aR1);
        aZ0 = wmma_f16(a, fragB(WzB + offA + kk, lane), aZ0);
        aZ1 = wmma_f16(a, fragB(WzB + offB + kk, lane), aZ1);
      }

      // ---- nonlinearities; write r*h; keep z and h_old in registers
      float zA[8], zB[8], hoA[8], hoB[8];
#pragma unroll
      for (int i = 0; i < 8; ++i) {
        int m = i + mOff;
        float r0 = fsig(aR0[i] + brA);
        float r1 = fsig(aR1[i] + brB);
        zA[i] = fsig(aZ0[i] + bzA);
        zB[i] = fsig(aZ1[i] + bzB);
        float h0 = (float)comb[m * DTOT + INPD + nA];
        float h1 = (float)comb[m * DTOT + INPD + nB];
        hoA[i] = h0;
        hoB[i] = h1;
        rh[m * HID + nA] = (_Float16)(r0 * h0);
        rh[m * HID + nB] = (_Float16)(r1 * h1);
      }
      __syncthreads();

      const _Float16* WhB = WhT + opaque_zero();

      // ---- candidate: [x | r*h] @ Wh
      v8f aH0 = {}, aH1 = {};
#pragma unroll
      for (int kk = 0; kk < INPD; kk += 32) {  // x part
        v16h a = fragA(arow + kk, lane);
        aH0 = wmma_f16(a, fragB(WhB + offA + kk, lane), aH0);
        aH1 = wmma_f16(a, fragB(WhB + offB + kk, lane), aH1);
      }
#pragma unroll
      for (int kk = 0; kk < HID; kk += 32) {  // r*h part
        v16h a = fragA(arow2 + kk, lane);
        aH0 = wmma_f16(a, fragB(WhB + offA + INPD + kk, lane), aH0);
        aH1 = wmma_f16(a, fragB(WhB + offB + INPD + kk, lane), aH1);
      }

      // ---- h_new = z*h + (1-z)*tanh(cand); write back to h-half of comb
#pragma unroll
      for (int i = 0; i < 8; ++i) {
        int m = i + mOff;
        float hc0 = ftanh(aH0[i] + bhA);
        float hc1 = ftanh(aH1[i] + bhB);
        float hn0 = zA[i] * hoA[i] + (1.f - zA[i]) * hc0;
        float hn1 = zB[i] * hoB[i] + (1.f - zB[i]) * hc1;
        comb[m * DTOT + INPD + nA] = (_Float16)hn0;
        comb[m * DTOT + INPD + nB] = (_Float16)hn1;
      }
    }
    __syncthreads();
  }

  // ---- output projection: h_last @ Wo + bo
  v8f o0 = {}, o1 = {};
  const _Float16* bOa = WoT + (size_t)nA * HID;
  const _Float16* bOb = WoT + (size_t)nB * HID;
  const _Float16* arowH = comb + (lane & 15) * DTOT + INPD;
#pragma unroll
  for (int kk = 0; kk < HID; kk += 32) {
    v16h a = fragA(arowH + kk, lane);
    o0 = wmma_f16(a, fragB(bOa + kk, lane), o0);
    o1 = wmma_f16(a, fragB(bOb + kk, lane), o1);
  }
  const float boA = bo[nA], boB = bo[nB];
#pragma unroll
  for (int i = 0; i < 8; ++i) {
    int m = i + mOff;
    out[(size_t)(bBase + m) * HID + nA] = o0[i] + boA;
    out[(size_t)(bBase + m) * HID + nB] = o1[i] + boB;
  }
}

// ---------------------------------------------------------------------------
extern "C" void kernel_launch(void* const* d_in, const int* in_sizes, int n_in,
                              void* d_out, int out_size, void* d_ws,
                              size_t ws_size, hipStream_t stream) {
  const float* x  = (const float*)d_in[0];
  const float* Wr = (const float*)d_in[1];
  const float* br = (const float*)d_in[2];
  const float* Wz = (const float*)d_in[3];
  const float* bz = (const float*)d_in[4];
  const float* Wh = (const float*)d_in[5];
  const float* bh = (const float*)d_in[6];
  const float* Wo = (const float*)d_in[7];
  const float* bo = (const float*)d_in[8];
  _Float16* Wt = (_Float16*)d_ws;  // 3*512*256 + 256*256 halves = 896 KB
  float* out = (float*)d_out;

  const int packElems = 3 * DTOT * HID + HID * HID;  // 458752
  gru_pack_weights<<<(packElems + 255) / 256, 256, 0, stream>>>(Wr, Wz, Wh, Wo, Wt);
  gru_seq_kernel<<<BATCH / BT, 256, 0, stream>>>(x, br, bz, bh, bo, Wt, out);
}
